// CustomLSTM2_9440338117186
// MI455X (gfx1250) — matrix operand order repaired and checked by reading, not compiled, not run on hardware
//
#include <hip/hip_runtime.h>
#include <hip/hip_bf16.h>
#include <cstdint>

typedef __attribute__((ext_vector_type(16))) __bf16 v16bf;
typedef __attribute__((ext_vector_type(2)))  __bf16 bf2;
typedef __attribute__((ext_vector_type(8)))  float  v8f;
typedef __attribute__((ext_vector_type(2)))  float  f2;

#define Bb 64
#define Ss 512
#define Dd 512
#define Hh 512
#define RS 514   // LDS h-buffer row stride in bf16 elems (spreads banks, keeps 4B align)

// ---------------------------------------------------------------------------
// Kernel 1: parallel precompute of layer-1 input projections
//   P1[t] = [W12; dW12] @ x_t^T  -> (16,64) f32 per t
// One workgroup per t, 4 waves, wave w computes batch tile w (16 cols), K=512.
// ---------------------------------------------------------------------------
__global__ __launch_bounds__(128)
void xproj_kernel(const float* __restrict__ x,
                  const float* __restrict__ W12,
                  const float* __restrict__ dW12,
                  float* __restrict__ P1) {
    const int t    = blockIdx.x;
    const int wave = threadIdx.x >> 5;     // 0..3 -> N tile
    const int lane = threadIdx.x & 31;
    const int l16  = lane & 15;
    const int hi   = lane >> 4;
    const int bcol = wave * 16 + l16;      // batch column for B / C fragments
    const float* arow = (l16 < 8) ? (W12 + l16 * Dd) : (dW12 + (l16 - 8) * Dd);

    v8f acc = {};
    for (int kb = 0; kb < Dd; kb += 32) {
        v16bf af, bf;
#pragma unroll
        for (int j = 0; j < 8; ++j) {
            // A (16x32, 16-bit) layout: lanes0-15 K=0..7,16..23 ; lanes16-31 K=8..15,24..31
            int Ka = (j < 4) ? (8 * hi + 2 * j) : (16 + 8 * hi + 2 * (j - 4));
            f2 a = *(const f2*)(arow + kb + Ka);
            af[2 * j]     = (__bf16)a[0];
            af[2 * j + 1] = (__bf16)a[1];
            // B (32x16) layout: lanes0-15 K=0..7,16..23 ; lanes16-31 K=8..15,24..31 ; N=lane%16
            int Kb = (j < 4) ? (8 * hi + 2 * j) : (16 + 8 * hi + 2 * (j - 4));
            f2 b = *(const f2*)(x + (size_t)bcol * (Ss * Dd) + (size_t)t * Dd + kb + Kb);
            bf[2 * j]     = (__bf16)b[0];
            bf[2 * j + 1] = (__bf16)b[1];
        }
        acc = __builtin_amdgcn_wmma_f32_16x16x32_bf16(false, af, false, bf,
                                                      (short)0, acc, false, false);
    }
    // C (16x16 f32): vgpr r -> M = r + 8*hi, N = lane%16
#pragma unroll
    for (int r = 0; r < 8; ++r) {
        int m = r + 8 * hi;
        P1[(size_t)t * 1024 + m * 64 + bcol] = acc[r];
    }
}

// ---------------------------------------------------------------------------
// Kernel 2: sequential recurrence, ONE workgroup of 512 threads (16 waves).
// h states live in LDS as bf16 [batch][hidden] (stride RS).
// c states live in d_ws (L2 resident).
// Per step: 3 GEMMs (16,512)@(512,64) via wmma bf16, then Kronecker-expand
// elementwise LSTM update.
// ---------------------------------------------------------------------------
__global__ __launch_bounds__(512)
void lstm_seq_kernel(const float* __restrict__ h0, const float* __restrict__ c0,
                     // layer 1
                     const float* __restrict__ W11, const float* __restrict__ dW11,
                     const float* __restrict__ U11, const float* __restrict__ dU11,
                     const float* __restrict__ U12, const float* __restrict__ dU12,
                     const float* __restrict__ b11, const float* __restrict__ b12,
                     // layer 2
                     const float* __restrict__ W21, const float* __restrict__ dW21,
                     const float* __restrict__ U21, const float* __restrict__ dU21,
                     const float* __restrict__ W22, const float* __restrict__ dW22,
                     const float* __restrict__ U22, const float* __restrict__ dU22,
                     const float* __restrict__ b21, const float* __restrict__ b22,
                     const float* __restrict__ P1,   // precomputed (S,16,64)
                     float* __restrict__ C1, float* __restrict__ C2,  // c state [j*64+b]
                     float* __restrict__ out) {
    __shared__ __bf16 Hb1[Bb * RS];
    __shared__ __bf16 Hb2[Bb * RS];
    __shared__ float  G0[16 * 64];
    __shared__ float  G1[16 * 64];
    __shared__ float  G2[16 * 64];
    __shared__ float  Part[16 * 256];

    const int tid  = threadIdx.x;
    const int wave = tid >> 5;           // 0..15
    const int lane = tid & 31;
    const int l16  = lane & 15;
    const int hi   = lane >> 4;
    const int ntile = wave & 3;          // output N tile (batch group of 16)
    const int kq    = wave >> 2;         // K quarter (128 wide)

    // ---- init h (from h0) and c (from c0) ----
    for (int e = tid; e < Bb * Hh; e += 512) {
        int b = e >> 9, j = e & 511;
        Hb1[b * RS + j] = (__bf16)h0[0 * Bb * Hh + b * Hh + j];
        Hb2[b * RS + j] = (__bf16)h0[1 * Bb * Hh + b * Hh + j];
    }
    for (int e = tid; e < Bb * Hh; e += 512) {
        int b = e & 63, j = e >> 6;
        C1[e] = c0[0 * Bb * Hh + b * Hh + j];
        C2[e] = c0[1 * Bb * Hh + b * Hh + j];
    }

    // ---- preload constant A fragments (weights) into VGPRs, bf16 ----
    // GEMM 0: [U12;dU12]  GEMM 1: [W22;dW22]  GEMM 2: [U22;dU22]
    v16bf AF[3][4];
    {
        const float* r0[3] = {U12, W22, U22};
        const float* r1[3] = {dU12, dW22, dU22};
#pragma unroll
        for (int g = 0; g < 3; ++g) {
            const float* arow = (l16 < 8) ? (r0[g] + l16 * Hh) : (r1[g] + (l16 - 8) * Hh);
#pragma unroll
            for (int w = 0; w < 4; ++w) {
                int kb = kq * 128 + w * 32;
#pragma unroll
                for (int j = 0; j < 8; ++j) {
                    int Ka = (j < 4) ? (8 * hi + 2 * j) : (16 + 8 * hi + 2 * (j - 4));
                    f2 a = *(const f2*)(arow + kb + Ka);
                    AF[g][w][2 * j]     = (__bf16)a[0];
                    AF[g][w][2 * j + 1] = (__bf16)a[1];
                }
            }
        }
    }
    __syncthreads();

    // GEMM: (16,512)@(512,64) -> Gout(16x64 f32 in LDS).
    // 16 waves = 4 N-tiles x 4 K-quarters; partial C reduced through LDS.
    auto gemm16 = [&](const v16bf (&Afr)[4], const __bf16* Hb, float* Gout) {
        v8f acc = {};
#pragma unroll
        for (int w = 0; w < 4; ++w) {
            int kb = kq * 128 + w * 32;
            v16bf bf;
#pragma unroll
            for (int j = 0; j < 8; ++j) {
                bf2 p = *(const bf2*)(Hb + (ntile * 16 + l16) * RS + kb + ((j < 4) ? (8 * hi + 2 * j) : (16 + 8 * hi + 2 * (j - 4))));
                bf[2 * j]     = p[0];
                bf[2 * j + 1] = p[1];
            }
            acc = __builtin_amdgcn_wmma_f32_16x16x32_bf16(false, Afr[w], false, bf,
                                                          (short)0, acc, false, false);
        }
#pragma unroll
        for (int r = 0; r < 8; ++r)
            Part[wave * 256 + (r + 8 * hi) * 16 + l16] = acc[r];
        __syncthreads();
        // reduce 4 K-partials; 1024 outputs over 512 threads
#pragma unroll
        for (int e0 = 0; e0 < 2; ++e0) {
            int e = tid + e0 * 512;
            int m = e >> 6, n = e & 63;
            int nt = n >> 4, nn = n & 15;
            float s = Part[(0 * 4 + nt) * 256 + m * 16 + nn]
                    + Part[(1 * 4 + nt) * 256 + m * 16 + nn]
                    + Part[(2 * 4 + nt) * 256 + m * 16 + nn]
                    + Part[(3 * 4 + nt) * 256 + m * 16 + nn];
            Gout[e] = s;
        }
        __syncthreads();
    };

    // Elementwise LSTM cell with Kronecker expansion.
    // Px rows 0-7 = W-path (8,64), rows 8-15 = dW-path; Gu likewise for U-path.
    auto cell = [&](const float* Px, const float* Gu,
                    const float* a1, const float* da1,
                    const float* u1, const float* du1,
                    const float* bb1, const float* bb2,
                    float* Cst, __bf16* HbOut,
                    float* seqOut, float* hnOut, float* cnOut) {
        const int b     = tid & 63;
        const int jbase = (tid >> 6) * 64;
        for (int jj = 0; jj < 64; ++jj) {
            int j  = jbase + jj;
            int fi = j & 7;        // Kronecker column factor index
            int gr = j >> 3;       // row within each gate's 64 G-groups
            float pvx  = Px[fi * 64 + b];
            float pvdx = Px[(8 + fi) * 64 + b];
            float uvx  = Gu[fi * 64 + b];
            float uvdx = Gu[(8 + fi) * 64 + b];
            float g4[4];
#pragma unroll
            for (int q = 0; q < 4; ++q) {
                int gi = q * 64 + gr;       // wave-uniform -> scalar loads
                int r  = gi * 8 + fi;
                g4[q] = a1[gi] * pvx + da1[gi] * pvdx
                      + u1[gi] * uvx + du1[gi] * uvdx + bb1[r] + bb2[r];
            }
            float ig = 1.f / (1.f + __expf(-g4[0]));
            float fg = 1.f / (1.f + __expf(-g4[1]));
            float gg = tanhf(g4[2]);
            float og = 1.f / (1.f + __expf(-g4[3]));
            float c  = fg * Cst[j * 64 + b] + ig * gg;
            float h  = og * tanhf(c);
            Cst[j * 64 + b] = c;
            HbOut[b * RS + j] = (__bf16)h;
            if (seqOut) seqOut[(size_t)b * (Ss * Hh) + j] = h;
            if (hnOut) { hnOut[b * Hh + j] = h; cnOut[b * Hh + j] = c; }
        }
    };

    const size_t OUT_HN = (size_t)Bb * Ss * Hh;
    const size_t OUT_CN = OUT_HN + 2 * (size_t)Bb * Hh;

    for (int t = 0; t < Ss; ++t) {
        bool last = (t == Ss - 1);
        // ---- layer 1 ----
        gemm16(AF[0], Hb1, G0);
        cell(P1 + (size_t)t * 1024, G0, W11, dW11, U11, dU11, b11, b12,
             C1, Hb1, nullptr,
             last ? (out + OUT_HN) : nullptr,
             last ? (out + OUT_CN) : nullptr);
        __syncthreads();
        // ---- layer 2 ----
        gemm16(AF[1], Hb1, G1);   // input projection from h1(t)
        gemm16(AF[2], Hb2, G2);   // recurrent projection from h2(t-1)
        cell(G1, G2, W21, dW21, U21, dU21, b21, b22,
             C2, Hb2, out + (size_t)t * Hh,
             last ? (out + OUT_HN + Bb * Hh) : nullptr,
             last ? (out + OUT_CN + Bb * Hh) : nullptr);
        __syncthreads();
    }
}

// ---------------------------------------------------------------------------
extern "C" void kernel_launch(void* const* d_in, const int* in_sizes, int n_in,
                              void* d_out, int out_size, void* d_ws, size_t ws_size,
                              hipStream_t stream) {
    const float* x    = (const float*)d_in[0];
    const float* h0   = (const float*)d_in[1];
    const float* c0   = (const float*)d_in[2];
    const float* W11  = (const float*)d_in[3];
    const float* W12  = (const float*)d_in[4];
    const float* U11  = (const float*)d_in[5];
    const float* U12  = (const float*)d_in[6];
    const float* dW11 = (const float*)d_in[7];
    const float* dW12 = (const float*)d_in[8];
    const float* dU11 = (const float*)d_in[9];
    const float* dU12 = (const float*)d_in[10];
    const float* b11  = (const float*)d_in[11];
    const float* b12  = (const float*)d_in[12];
    const float* W21  = (const float*)d_in[13];
    const float* W22  = (const float*)d_in[14];
    const float* U21  = (const float*)d_in[15];
    const float* U22  = (const float*)d_in[16];
    const float* dW21 = (const float*)d_in[17];
    const float* dW22 = (const float*)d_in[18];
    const float* dU21 = (const float*)d_in[19];
    const float* dU22 = (const float*)d_in[20];
    const float* b21  = (const float*)d_in[21];
    const float* b22  = (const float*)d_in[22];

    float* P1 = (float*)d_ws;                  // S*16*64 f32 = 2 MB
    float* C1 = P1 + (size_t)Ss * 1024;        // 64*512 f32
    float* C2 = C1 + (size_t)Bb * Hh;          // 64*512 f32
    float* out = (float*)d_out;

    // Phase 1: parallel input projections for all timesteps
    xproj_kernel<<<Ss, 128, 0, stream>>>(x, W12, dW12, P1);

    // Phase 2: sequential recurrence, single WGP-resident workgroup
    lstm_seq_kernel<<<1, 512, 0, stream>>>(h0, c0,
                                           W11, dW11, U11, dU11, U12, dU12, b11, b12,
                                           W21, dW21, U21, dU21, W22, dW22, U22, dU22,
                                           b21, b22,
                                           P1, C1, C2, out);
}
